// Network_63273458205287
// MI455X (gfx1250) — matrix operand order repaired and checked by reading, not compiled, not run on hardware
//
#include <hip/hip_runtime.h>
#include <hip/hip_bf16.h>

typedef __attribute__((ext_vector_type(16))) _Float16 v16h;
typedef __attribute__((ext_vector_type(8)))  float    v8f;

// ---------------------------------------------------------------------------
// Small utility kernels
// ---------------------------------------------------------------------------
__global__ void zero_f32(float* __restrict__ p, long long n) {
    long long i = (long long)blockIdx.x * blockDim.x + threadIdx.x;
    long long stride = (long long)gridDim.x * blockDim.x;
    for (; i < n; i += stride) p[i] = 0.0f;
}

__global__ void fill_f32(float* __restrict__ p, float v, int n) {
    int i = blockIdx.x * blockDim.x + threadIdx.x;
    if (i < n) p[i] = v;
}

// degree accumulation: deg[row[e]] += 1 (deg pre-filled with 1.0 for self-loop)
__global__ void deg_kernel(const int* __restrict__ row, float* __restrict__ deg, int E) {
    int e = blockIdx.x * blockDim.x + threadIdx.x;
    if (e < E) atomicAdd(&deg[row[e]], 1.0f);
}

__global__ void rsqrt_kernel(float* __restrict__ d, int n) {
    int i = blockIdx.x * blockDim.x + threadIdx.x;
    if (i < n) d[i] = rsqrtf(d[i]);
}

// ---------------------------------------------------------------------------
// Pack weight matrix W [K x N] (f32, row-major) into the per-lane v16h layout
// required by v_wmma_f32_16x16x32_f16's B operand (32x16 K-major fragments).
// Layout: pack[((ks*ntiles + nt)*32 + lane)*16 + 2j (+1)]
//   lanes 0-15 : K = ks*32 + 2j (j<4) or 2j+8 (j>=4),    N = nt*16 + lane
//   lanes 16-31: K = ks*32 + 2j+8 (j<4) or 2j+16 (j>=4), N = nt*16 + (lane-16)
// ---------------------------------------------------------------------------
__global__ void pack_B(const float* __restrict__ W, _Float16* __restrict__ pack,
                       int N, int ksteps, int ntiles) {
    int total = ksteps * ntiles * 32 * 8;
    for (int i = blockIdx.x * blockDim.x + threadIdx.x; i < total;
         i += gridDim.x * blockDim.x) {
        int j    = i & 7;
        int lane = (i >> 3) & 31;
        int t    = i >> 8;              // ks*ntiles + nt
        int nt   = t % ntiles;
        int ks   = t / ntiles;
        int hi   = lane >> 4;
        int k    = ks * 32 + ((j < 4) ? 2 * j : 2 * j + 8) + (hi ? 8 : 0);
        int n    = nt * 16 + (lane & 15);
        size_t o = (size_t)t * 512 + (size_t)lane * 16 + 2 * j;
        pack[o]     = (_Float16)W[(size_t)k * N + n];
        pack[o + 1] = (_Float16)W[(size_t)(k + 1) * N + n];
    }
}

// ---------------------------------------------------------------------------
// WMMA GEMM: C[M x N] = A[M x K] (f32, converted to f16 on the fly) * Bpack.
// One wave per 16x16 output tile, K walked in steps of 32.
// ---------------------------------------------------------------------------
__global__ void gcn_wmma_gemm(const float* __restrict__ A,
                              const _Float16* __restrict__ Bpack,
                              float* __restrict__ C,
                              int M, int K, int N, int ksteps, int ntilesN,
                              int totalTiles) {
    int wave = blockIdx.x * (blockDim.x >> 5) + (threadIdx.x >> 5);
    if (wave >= totalTiles) return;           // whole-wave uniform: EXEC all-1s inside
    int lane  = threadIdx.x & 31;
    int tileN = wave % ntilesN;
    int tileM = wave / ntilesN;
    int hi    = lane >> 4;                    // half-wave selector
    int mrow  = tileM * 16 + (lane & 15);
    if (mrow > M - 1) mrow = M - 1;           // clamp (keeps EXEC uniform)
    size_t rowA = (size_t)mrow * K;

    v8f c = {};
    for (int ks = 0; ks < ksteps; ++ks) {
        // Build A fragment per ISA 16-bit A 16x32 layout (f32 -> f16 cvt on the fly)
        v16h a;
        int k0 = ks * 32 + (hi ? 8 : 0);
#pragma unroll
        for (int j = 0; j < 8; ++j) {
            int k = k0 + ((j < 4) ? 2 * j : 2 * j + 8);
            float2 xv = *(const float2*)(A + rowA + k);
            a[2 * j]     = (_Float16)xv.x;
            a[2 * j + 1] = (_Float16)xv.y;
        }
        // B fragment: one contiguous 32B load per lane
        v16h b = *(const v16h*)(Bpack +
                   ((size_t)(ks * ntilesN + tileN) * 32 + lane) * 16);
        c = __builtin_amdgcn_wmma_f32_16x16x32_f16(
                false, a, false, b, (short)0, c, false, false);
    }

    // Store D per 32-bit C/D 16x16 layout
    int colBase = tileN * 16 + (lane & 15);
    int rowBase = tileM * 16 + hi * 8;
#pragma unroll
    for (int r = 0; r < 8; ++r) {
        int rr = rowBase + r;
        if (rr < M) C[(size_t)rr * N + colBase] = c[r];
    }
}

// ---------------------------------------------------------------------------
// Edge aggregation: out[row[e], f] += H[col[e], f] * dinv[row]*dinv[col]
// One thread per (edge, feature); feature index contiguous -> coalesced rows,
// scatter via global_atomic_add_f32 (L2-resident: H fits in 192MB L2).
// ---------------------------------------------------------------------------
__global__ void agg_scatter(const int* __restrict__ row, const int* __restrict__ col,
                            const float* __restrict__ dinv, const float* __restrict__ H,
                            float* __restrict__ out, int E, int F) {
    long long idx = (long long)blockIdx.x * blockDim.x + threadIdx.x;
    long long total = (long long)E * F;
    if (idx >= total) return;
    int f = (int)(idx & (F - 1));     // F is 64 or 32 (power of two)
    int e = (int)(idx / F);
    int r = row[e];
    int cn = col[e];
    float nrm = dinv[r] * dinv[cn];
    atomicAdd(&out[(size_t)r * F + f], H[(size_t)cn * F + f] * nrm);
}

// agg += self-loop term + bias, then ReLU (in place)
__global__ void bias_relu_selfloop(float* __restrict__ agg, const float* __restrict__ H,
                                   const float* __restrict__ dinv,
                                   const float* __restrict__ b, int Nn, int F) {
    long long idx = (long long)blockIdx.x * blockDim.x + threadIdx.x;
    long long total = (long long)Nn * F;
    if (idx >= total) return;
    int f = (int)(idx & (F - 1));
    int i = (int)(idx / F);
    float di = dinv[i];
    float v = agg[idx] + H[idx] * di * di + b[f];
    agg[idx] = fmaxf(v, 0.0f);
}

// Final: out += self-loop + bias, then log_softmax over 32 classes.
// One wave32 per node (lane == class), shuffle reductions.
__global__ void final_logsoftmax(float* __restrict__ out, const float* __restrict__ H2,
                                 const float* __restrict__ dinv,
                                 const float* __restrict__ b, int Nn) {
    int node = blockIdx.x * (blockDim.x >> 5) + (threadIdx.x >> 5);
    if (node >= Nn) return;
    int lane = threadIdx.x & 31;
    size_t idx = (size_t)node * 32 + lane;
    float di = dinv[node];
    float v = out[idx] + H2[idx] * di * di + b[lane];
    float m = v;
#pragma unroll
    for (int o = 16; o > 0; o >>= 1) m = fmaxf(m, __shfl_xor(m, o, 32));
    float ex = expf(v - m);
    float s = ex;
#pragma unroll
    for (int o = 16; o > 0; o >>= 1) s += __shfl_xor(s, o, 32);
    out[idx] = v - m - logf(s);
}

// ---------------------------------------------------------------------------
// Launch: x[100000x128] f32, edge_index[2x1.6M] i32, W1[128x64], b1[64],
//         W2[64x32], b2[32] -> out[100000x32] (log-probs)
// ---------------------------------------------------------------------------
extern "C" void kernel_launch(void* const* d_in, const int* in_sizes, int n_in,
                              void* d_out, int out_size, void* d_ws, size_t ws_size,
                              hipStream_t stream) {
    const float* x  = (const float*)d_in[0];
    const int*   ei = (const int*)d_in[1];
    const float* W1 = (const float*)d_in[2];
    const float* b1 = (const float*)d_in[3];
    const float* W2 = (const float*)d_in[4];
    const float* b2 = (const float*)d_in[5];
    float* out = (float*)d_out;

    const int F_IN = 128, F_H = 64, F_O = 32;
    int Nn = in_sizes[0] / F_IN;          // 100000
    int E  = in_sizes[1] / 2;             // 1600000
    const int* row = ei;
    const int* col = ei + E;

    // workspace carve-out (all offsets are 32B-aligned multiples)
    float* h1    = (float*)d_ws;                    // Nn * 64
    float* agg1  = h1 + (size_t)Nn * F_H;           // Nn * 64 (becomes ReLU activations)
    float* h2    = agg1 + (size_t)Nn * F_H;         // Nn * 32
    float* dinv  = h2 + (size_t)Nn * F_O;           // Nn
    _Float16* pack1 = (_Float16*)(dinv + Nn);       // 4*4*512 = 8192 halves
    _Float16* pack2 = pack1 + 8192;                 // 2*2*512 = 2048 halves

    const int B = 256;
    long long nAgg1 = (long long)Nn * F_H;

    // init
    zero_f32<<<4096, B, 0, stream>>>(agg1, nAgg1);
    zero_f32<<<4096, B, 0, stream>>>(out, (long long)Nn * F_O);
    fill_f32<<<(Nn + B - 1) / B, B, 0, stream>>>(dinv, 1.0f, Nn);   // self-loop deg

    // degrees -> dinv
    deg_kernel<<<(E + B - 1) / B, B, 0, stream>>>(row, dinv, E);
    rsqrt_kernel<<<(Nn + B - 1) / B, B, 0, stream>>>(dinv, Nn);

    // pack weights into WMMA B-operand layout
    pack_B<<<32, B, 0, stream>>>(W1, pack1, F_H, /*ksteps=*/4, /*ntiles=*/4);
    pack_B<<<8,  B, 0, stream>>>(W2, pack2, F_O, /*ksteps=*/2, /*ntiles=*/2);

    // layer 1: h1 = x @ W1 (WMMA f16->f32)
    int mt = (Nn + 15) / 16;
    int tiles1 = mt * 4;
    gcn_wmma_gemm<<<(tiles1 + 7) / 8, B, 0, stream>>>(
        x, pack1, h1, Nn, F_IN, F_H, 4, 4, tiles1);

    // layer 1 aggregation + bias + relu (self-loops folded analytically)
    long long w1work = (long long)E * F_H;
    agg_scatter<<<(unsigned)((w1work + B - 1) / B), B, 0, stream>>>(
        row, col, dinv, h1, agg1, E, F_H);
    bias_relu_selfloop<<<(unsigned)((nAgg1 + B - 1) / B), B, 0, stream>>>(
        agg1, h1, dinv, b1, Nn, F_H);

    // layer 2: h2 = relu_out @ W2 (WMMA f16->f32)
    int tiles2 = mt * 2;
    gcn_wmma_gemm<<<(tiles2 + 7) / 8, B, 0, stream>>>(
        agg1, pack2, h2, Nn, F_H, F_O, 2, 2, tiles2);

    // layer 2 aggregation into d_out, then self-loop + bias + log_softmax
    long long w2work = (long long)E * F_O;
    agg_scatter<<<(unsigned)((w2work + B - 1) / B), B, 0, stream>>>(
        row, col, dinv, h2, out, E, F_O);
    final_logsoftmax<<<(Nn + 7) / 8, B, 0, stream>>>(out, h2, dinv, b2, Nn);
}
